// PairwiseActor_53377853555233
// MI455X (gfx1250) — compile-verified, matrix operand order checked by reading, not run on hardware
//
#include <hip/hip_runtime.h>

typedef __attribute__((ext_vector_type(16))) _Float16 v16h;
typedef __attribute__((ext_vector_type(8)))  _Float16 v8h;
typedef __attribute__((ext_vector_type(4)))  _Float16 v4h;
typedef __attribute__((ext_vector_type(8)))  float    v8f;

#define B_ 8
#define U_ 128
#define T_ 256
#define E_ 128
#define H_ 128
#define KPAD 8
#define WAVES 2
#define MT 32            // t-rows per wave (two 16-row WMMA tiles)

// Branchless GELU: 0.5x(1+erf(x/sqrt2)), A&S 7.1.25 erf (|err| <= 2.5e-5,
// far below f16-WMMA quantization). ~11 straight-line VALU/TRANS ops.
__device__ __forceinline__ float gelu_fast(float x) {
    const float ax = __builtin_fabsf(x) * 0.70710678118654752440f;
    const float t  = __builtin_amdgcn_rcpf(fmaf(0.47047f, ax, 1.0f));
    float p = fmaf(0.7478556f, t, -0.0958798f);
    p = fmaf(p, t, 0.3480242f);
    p = p * t;
    const float e  = __expf(-ax * ax);
    const float er = fmaf(-p, e, 1.0f);                  // erf(|x|/sqrt2)
    const float se = __builtin_copysignf(er, x);
    const float hx = 0.5f * x;
    return fmaf(hx, se, hx);
}

// ---------------------------------------------------------------------------
// Prep 1: rank-separable layer-1 partials (fp32).
// ---------------------------------------------------------------------------
__global__ __launch_bounds__(128) void partials_kernel(
    const float* __restrict__ ue, const float* __restrict__ te,
    const float* __restrict__ ge, const float* __restrict__ W1,
    const float* __restrict__ b1,
    float* __restrict__ au, float* __restrict__ at, float* __restrict__ ag)
{
    __shared__ float sx[E_];
    const int r = blockIdx.x;
    const int h = threadIdx.x;
    const float* x; const float* W; float* o; float acc;
    if (r < B_*U_) {
        x = ue + (size_t)r*E_; W = W1; o = au + (size_t)r*H_; acc = 0.f;
    } else if (r < B_*U_ + B_*T_) {
        const int i = r - B_*U_;
        x = te + (size_t)i*E_; W = W1 + (size_t)E_*H_; o = at + (size_t)i*H_; acc = 0.f;
    } else {
        const int i = r - B_*U_ - B_*T_;
        x = ge + (size_t)i*E_; W = W1 + (size_t)2*E_*H_; o = ag + (size_t)i*H_; acc = b1[h];
    }
    sx[h] = x[h];
    __syncthreads();
    #pragma unroll 8
    for (int e = 0; e < E_; ++e)
        acc = fmaf(sx[e], W[(size_t)e*H_ + h], acc);
    o[h] = acc;
}

// ---------------------------------------------------------------------------
// Prep 2: W2 fp32 -> f16, transposed ([n][k]) for contiguous b128 LDS staging.
// ---------------------------------------------------------------------------
__global__ __launch_bounds__(256) void convw2_kernel(
    const float* __restrict__ W2, _Float16* __restrict__ w2t)
{
    const int i = blockIdx.x * 256 + threadIdx.x;        // i = n*H + k
    if (i < H_*H_) {
        const int n = i >> 7, k = i & 127;
        w2t[i] = (_Float16)W2[(size_t)k*H_ + n];
    }
}

// ---------------------------------------------------------------------------
// Main fused kernel. One wave = one (b, u, 32 consecutive t) super-tile:
//   h1 = gelu(au + at + ag)        (32 x 128 f16 in LDS, b128-coalesced)
//   h2 = gelu(h1 @ W2 + b2)        (8 N x 4 K x 2 M = 64 v_wmma_f32_16x16x32_f16)
//   out = h2 @ w3 + b3             (shfl_xor half-wave reduction)
// ---------------------------------------------------------------------------
__global__ __launch_bounds__(WAVES*32) void pairwise_kernel(
    const float* __restrict__ au, const float* __restrict__ at,
    const float* __restrict__ ag, const _Float16* __restrict__ w2t,
    const float* __restrict__ b2, const float* __restrict__ w3,
    const float* __restrict__ b3, float* __restrict__ out)
{
    __shared__ __attribute__((aligned(16))) _Float16 sW2[H_][H_ + KPAD];      // [n][k]
    __shared__ __attribute__((aligned(16))) _Float16 sH1[WAVES][MT][H_ + KPAD];

    const int tid  = threadIdx.x;
    const int w    = tid >> 5;
    const int lane = tid & 31;

    const int g    = blockIdx.x * WAVES + w;
    const int nTt  = T_ / MT;                 // 8 super-tiles along t
    const int tt   = g % nTt;
    const int u    = (g / nTt) % U_;
    const int b    = g / (nTt * U_);
    const int t0   = tt * MT;

    // ---- stage W2^T into LDS: 16B vector copies, padded rows ----
    for (int i = tid; i < H_*16; i += WAVES*32) {
        const int row = i >> 4, c = i & 15;
        ((uint4*)&sW2[row][0])[c] = ((const uint4*)(w2t + (size_t)row*H_))[c];
    }

    // ---- hoist epilogue constants (issue loads early, once) ----
    const int m = lane & 15;
    float biasv[8], wvv[8];
    #pragma unroll
    for (int nt = 0; nt < 8; ++nt) {
        biasv[nt] = b2[nt*16 + m];
        wvv[nt]   = w3[nt*16 + m];
    }
    const float b3v = b3[0];

    // ---- registerize the (b,u)-broadcast partials, 4 contiguous k per lane ----
    const float* aup = au + (size_t)(b*U_ + u)*H_;
    const float* agp = ag + (size_t)b*H_;
    float4 augv;
    {
        const float4 a4 = ((const float4*)aup)[lane];
        const float4 g4 = ((const float4*)agp)[lane];
        augv.x = a4.x + g4.x; augv.y = a4.y + g4.y;
        augv.z = a4.z + g4.z; augv.w = a4.w + g4.w;
    }

    // ---- stage h1 = gelu(aug + at) as f16: one b128 load + b64 LDS store/row ----
    const float* atp = at + (size_t)(b*T_ + t0)*H_;
    #pragma unroll 4
    for (int tl = 0; tl < MT; ++tl) {
        const float4 v = ((const float4*)(atp + (size_t)tl*H_))[lane];
        v4h hv;
        hv[0] = (_Float16)gelu_fast(v.x + augv.x);
        hv[1] = (_Float16)gelu_fast(v.y + augv.y);
        hv[2] = (_Float16)gelu_fast(v.z + augv.z);
        hv[3] = (_Float16)gelu_fast(v.w + augv.w);
        *(v4h*)&sH1[w][tl][lane*4] = hv;
    }
    __syncthreads();

    // ---- A operands (16x32 f16 per K-step, ISA lane layout) ----
    // lanes 0-15: row M=lane,    K = kt*32 + {0..7, 16..23}
    // lanes 16-31: row M=lane-16, K = kt*32 + {8..15, 24..31}
    const int off = (lane < 16) ? 0 : 8;
    v16h a[2][4];
    #pragma unroll
    for (int mt = 0; mt < 2; ++mt) {
        #pragma unroll
        for (int kt = 0; kt < 4; ++kt) {
            const _Float16* rowp = &sH1[w][mt*16 + m][kt*32 + off];
            v16h tmp;
            ((v8h*)&tmp)[0] = *(const v8h*)(rowp);
            ((v8h*)&tmp)[1] = *(const v8h*)(rowp + 16);
            a[mt][kt] = tmp;
        }
    }

    float s[2][8];
    #pragma unroll
    for (int mt = 0; mt < 2; ++mt)
        #pragma unroll
        for (int r = 0; r < 8; ++r) s[mt][r] = 0.f;

    // ---- GEMM vs W2 with fused bias/GELU/W3 epilogue ----
    #pragma unroll
    for (int nt = 0; nt < 8; ++nt) {
        const int n = nt*16 + m;              // D layout: lane%16 == column N
        v8f acc0 = {}, acc1 = {};
        #pragma unroll
        for (int kt = 0; kt < 4; ++kt) {
            const _Float16* colp = &sW2[n][kt*32 + off];
            v16h bop;
            ((v8h*)&bop)[0] = *(const v8h*)(colp);
            ((v8h*)&bop)[1] = *(const v8h*)(colp + 16);
            acc0 = __builtin_amdgcn_wmma_f32_16x16x32_f16(
                     false, a[0][kt], false, bop, (short)0, acc0, false, false);
            acc1 = __builtin_amdgcn_wmma_f32_16x16x32_f16(
                     false, a[1][kt], false, bop, (short)0, acc1, false, false);
        }
        const float bias = biasv[nt];
        const float wv   = wvv[nt];
        #pragma unroll
        for (int r = 0; r < 8; ++r) {         // VGPR slot r: M=r (lo half) / r+8 (hi half)
            s[0][r] = fmaf(gelu_fast(acc0[r] + bias), wv, s[0][r]);
            s[1][r] = fmaf(gelu_fast(acc1[r] + bias), wv, s[1][r]);
        }
    }

    // ---- reduce across the 16 lanes of each half-wave (columns of h2) ----
    #pragma unroll
    for (int d = 8; d > 0; d >>= 1) {
        #pragma unroll
        for (int mt = 0; mt < 2; ++mt)
            #pragma unroll
            for (int r = 0; r < 8; ++r)
                s[mt][r] += __shfl_xor(s[mt][r], d, 32);
    }

    if ((lane & 15) == 0) {
        const int rbase = (lane < 16) ? 0 : 8;
        #pragma unroll
        for (int mt = 0; mt < 2; ++mt) {
            float* op = out + (size_t)b*U_*T_ + (size_t)u*T_ + t0 + mt*16 + rbase;
            #pragma unroll
            for (int r = 0; r < 8; ++r) op[r] = s[mt][r] + b3v;
        }
    }
}

// ---------------------------------------------------------------------------
extern "C" void kernel_launch(void* const* d_in, const int* in_sizes, int n_in,
                              void* d_out, int out_size, void* d_ws, size_t ws_size,
                              hipStream_t stream) {
    (void)in_sizes; (void)n_in; (void)out_size; (void)ws_size;

    const float* ue = (const float*)d_in[0];
    const float* te = (const float*)d_in[1];
    const float* ge = (const float*)d_in[2];
    const float* W1 = (const float*)d_in[3];
    const float* b1 = (const float*)d_in[4];
    const float* W2 = (const float*)d_in[5];
    const float* b2 = (const float*)d_in[6];
    const float* w3 = (const float*)d_in[7];
    const float* b3 = (const float*)d_in[8];
    float* out = (float*)d_out;

    // workspace carving (~1.6 MB total, stays L2-hot)
    float*    au  = (float*)d_ws;                       // B*U*H
    float*    at  = au + (size_t)B_*U_*H_;              // B*T*H
    float*    ag  = at + (size_t)B_*T_*H_;              // B*H
    _Float16* w2t = (_Float16*)(ag + (size_t)B_*H_);    // H*H halves (transposed)

    const int rows = B_*U_ + B_*T_ + B_;
    partials_kernel<<<rows, 128, 0, stream>>>(ue, te, ge, W1, b1, au, at, ag);
    convw2_kernel<<<(H_*H_ + 255)/256, 256, 0, stream>>>(W2, w2t);

    const int tiles = B_ * U_ * (T_/MT);                // 8192 waves
    pairwise_kernel<<<tiles/WAVES, WAVES*32, 0, stream>>>(
        au, at, ag, w2t, b2, w3, b3, out);
}
